// ModelEncoder_32744830664832
// MI455X (gfx1250) — matrix operand order repaired and checked by reading, not compiled, not run on hardware
//
#include <hip/hip_runtime.h>

typedef __attribute__((ext_vector_type(16))) _Float16 v16h;
typedef __attribute__((ext_vector_type(8)))  _Float16 v8h;
typedef __attribute__((ext_vector_type(8)))  float    v8f;

#define NN 50000
#define EE 800000
#define GG 512
#define HH 76
#define FSTRIDE 80   // padded per-row feature stride (f16 elements, 160B)

static __device__ __forceinline__ float relu_f(float v) { return v > 0.f ? v : 0.f; }

// ---------------------------------------------------------------------------
// Encoder: DIN=1 so Linear is an outer product. out[i][c] = relu(x[i]*w[c]+b[c])
// ---------------------------------------------------------------------------
__global__ void encode_kernel(const float* __restrict__ x,
                              const float* __restrict__ w,
                              const float* __restrict__ b,
                              _Float16* __restrict__ out, int n) {
  int t = blockIdx.x * blockDim.x + threadIdx.x;
  if (t >= n * 16) return;
  int i = t >> 4, c = t & 15;
  float v = x[i] * w[c] + b[c];
  out[(size_t)i * FSTRIDE + c] = (_Float16)relu_f(v);
}

// ---------------------------------------------------------------------------
// Repack f32 weight into f16 WMMA B-fragment layout with segment-padded K:
//   fragment index: [chunk][colTile][lane][j], 16 halves contiguous per lane
//   padded k = chunk*32 + (lane>=16?8:0) + (j&7) + (j>=8?16:0)
//   seg = k/segW, off = k%segW ; original row = seg*hseg + off  (valid if
//   seg<3 && off<hseg) ; zero otherwise.  (Non-segmented weights: segW=Kpad.)
// ---------------------------------------------------------------------------
__global__ void conv_wfrag(const float* __restrict__ W, _Float16* __restrict__ Wf,
                           int hseg, int Ncols, int Kpad, int NT, int segW) {
  int idx = blockIdx.x * blockDim.x + threadIdx.x;
  int total = (Kpad >> 5) * NT * 32 * 16;
  if (idx >= total) return;
  int j = idx & 15;
  int l = (idx >> 4) & 31;
  int rem = idx >> 9;
  int t = rem % NT;
  int c = rem / NT;
  int kp = (c << 5) + ((l & 16) ? 8 : 0) + (j & 7) + ((j & 8) ? 16 : 0);
  int seg = kp / segW, off = kp % segW;
  int n = (t << 4) + (l & 15);
  float v = 0.f;
  if (seg < 3 && off < hseg && n < Ncols)
    v = W[(size_t)(seg * hseg + off) * Ncols + n];
  Wf[idx] = (_Float16)v;
}

// ---------------------------------------------------------------------------
// Per-lane WMMA A fragment from LDS (16-bit 16x32 layout)
// ---------------------------------------------------------------------------
static __device__ __forceinline__ v16h load_a_frag(const _Float16* ap) {
  v8h lo  = *(const v8h*)ap;
  v8h hi8 = *(const v8h*)(ap + 16);
  v16h a;
#pragma unroll
  for (int j = 0; j < 8; ++j) { a[j] = lo[j]; a[j + 8] = hi8[j]; }
  return a;
}

// ---------------------------------------------------------------------------
// Edge message: m = relu([h[src], h[dst], e] @ W + bias); agg[dst] += m
// Block = 192 threads (6 waves) handling 64 edges (4 row tiles of 16).
// A tile gathered with async global->LDS B128 copies (segment-padded layout),
// B fragments register-resident across all 4 row tiles.
// ---------------------------------------------------------------------------
template <int KPAD, int SEGW>
__global__ __launch_bounds__(192) void edge_msg_wmma_async(
    const _Float16* __restrict__ hbuf, const _Float16* __restrict__ ebuf,
    const int* __restrict__ src, const int* __restrict__ dst,
    const v16h* __restrict__ Wfrag, const float* __restrict__ bias,
    _Float16* __restrict__ mbuf, float* __restrict__ agg, int nE) {
  constexpr int LROW = KPAD + 8;            // LDS row stride in halves (16B mult)
  constexpr int NBLK = SEGW >> 3;           // 16B blocks per segment
  constexpr int NC   = KPAD >> 5;           // WMMA K chunks
  __shared__ __align__(16) _Float16 Atile[64 * LROW];
  const int tid = threadIdx.x;
  const int e0 = blockIdx.x << 6;

  // ---- zero the per-row tail [3*SEGW, KPAD) : always 16 halves = 8 dwords ----
  for (int i = tid; i < 64 * 8; i += 192) {
    int row = i >> 3, w = i & 7;
    *(unsigned*)&Atile[row * LROW + 3 * SEGW + (w << 1)] = 0u;
  }

  // ---- async gather: 64 rows x 3 segments x NBLK 16-byte blocks ----
  for (int i = tid; i < 64 * 3 * NBLK; i += 192) {
    int b   = i % NBLK;
    int seg = (i / NBLK) % 3;
    int row = i / (3 * NBLK);
    int eidx = e0 + row; if (eidx >= nE) eidx = nE - 1;
    const _Float16* gsrc;
    if (seg == 0)      gsrc = hbuf + (size_t)src[eidx] * FSTRIDE;
    else if (seg == 1) gsrc = hbuf + (size_t)dst[eidx] * FSTRIDE;
    else               gsrc = ebuf + (size_t)eidx * FSTRIDE;
    unsigned long long gaddr = (unsigned long long)(const void*)(gsrc + (b << 3));
    unsigned lds_addr =
        (unsigned)(uintptr_t)&Atile[row * LROW + seg * SEGW + (b << 3)];
    asm volatile("global_load_async_to_lds_b128 %0, %1, off"
                 :: "v"(lds_addr), "v"(gaddr) : "memory");
  }
  asm volatile("s_wait_asynccnt 0x0" ::: "memory");
  __syncthreads();

  const int wave = tid >> 5;
  const int lane = tid & 31;
  const int col  = (wave << 4) + (lane & 15);
  const int arow = lane & 15;
  const int hi   = lane >> 4;

  // ---- B fragments: register resident across all 4 row tiles ----
  v16h breg[NC];
#pragma unroll
  for (int c = 0; c < NC; ++c)
    breg[c] = Wfrag[((size_t)c * 6 + wave) * 32 + lane];

  const float bv = (col < HH) ? bias[col] : 0.f;

#pragma unroll
  for (int rt = 0; rt < 4; ++rt) {
    const _Float16* ap0 = &Atile[((rt << 4) + arow) * LROW + hi * 8];
    v8f acc = {};
#pragma unroll
    for (int c = 0; c < NC; ++c) {
      v16h a = load_a_frag(ap0 + (c << 5));
      acc = __builtin_amdgcn_wmma_f32_16x16x32_f16(false, a, false, breg[c],
                                                   (short)0, acc, false, false);
    }
    if (col < FSTRIDE) {
      const int mbase = (rt << 4) + (hi << 3);
#pragma unroll
      for (int r = 0; r < 8; ++r) {
        int eidx = e0 + mbase + r;
        if (eidx < nE) {
          float v = relu_f(acc[r] + bv);
          mbuf[(size_t)eidx * FSTRIDE + col] = (_Float16)v;  // incl. zero padding cols
          if (col < HH)
            atomicAdd(&agg[(size_t)dst[eidx] * FSTRIDE + col], v);
        }
      }
    }
  }
}

// ---------------------------------------------------------------------------
// Node update: h_new = relu([h, agg] @ W + bias)   (cold path: 16-row tiles)
// ---------------------------------------------------------------------------
__global__ __launch_bounds__(192) void node_upd_wmma(
    const _Float16* __restrict__ hbuf, const float* __restrict__ agg,
    const v16h* __restrict__ Wfrag, const float* __restrict__ bias,
    _Float16* __restrict__ hnew, int nN, int hw, int Kpad) {
  __shared__ __align__(16) _Float16 Atile[16 * (160 + 8)];
  const int LDSK = Kpad + 8;
  const int tid = threadIdx.x;
  const int n0 = blockIdx.x << 4;
  const int Kin = hw + HH;

  {
    int row = tid & 15;
    int nidx = n0 + row;
    bool rv = nidx < nN;
    const _Float16* rh = hbuf + (size_t)(rv ? nidx : 0) * FSTRIDE;
    const float* ra = agg + (size_t)(rv ? nidx : 0) * FSTRIDE;
    for (int K = tid >> 4; K < Kpad; K += 12) {
      float v = 0.f;
      if (rv) {
        if (K < hw)       v = (float)rh[K];
        else if (K < Kin) v = ra[K - hw];
      }
      Atile[row * LDSK + K] = (_Float16)v;
    }
  }
  __syncthreads();

  const int wave = tid >> 5;
  const int lane = tid & 31;
  const int col  = (wave << 4) + (lane & 15);
  const int arow = lane & 15;
  const int hi   = lane >> 4;

  v8f acc = {};
  for (int k0 = 0; k0 < Kpad; k0 += 32) {
    v16h a = load_a_frag(&Atile[arow * LDSK + k0 + hi * 8]);
    v16h b = Wfrag[((size_t)(k0 >> 5) * 6 + wave) * 32 + lane];
    acc = __builtin_amdgcn_wmma_f32_16x16x32_f16(false, a, false, b,
                                                 (short)0, acc, false, false);
  }

  if (col < FSTRIDE) {
    float bv = (col < HH) ? bias[col] : 0.f;
    const int mbase = hi << 3;
#pragma unroll
    for (int r = 0; r < 8; ++r) {
      int nidx = n0 + mbase + r;
      if (nidx < nN)
        hnew[(size_t)nidx * FSTRIDE + col] = (_Float16)relu_f(acc[r] + bv);
    }
  }
}

// ---------------------------------------------------------------------------
// Edge output projection fused with per-graph pooled sum:
//   eo = m @ eow + eob ; sums[batch[src[e]]] += eo
// ---------------------------------------------------------------------------
__global__ __launch_bounds__(128) void edge_out_pool_wmma(
    const _Float16* __restrict__ mbuf, const v16h* __restrict__ Wfrag,
    const float* __restrict__ eob, const int* __restrict__ src,
    const int* __restrict__ batch, float* __restrict__ sums, int nE) {
  __shared__ __align__(16) _Float16 Atile[16 * (96 + 8)];
  const int Kpad = 96, LDSK = 96 + 8;
  const int tid = threadIdx.x;
  const int e0 = blockIdx.x << 4;

  {
    int row = tid & 15;
    int eidx = e0 + row;
    bool rv = eidx < nE;
    const _Float16* rm = mbuf + (size_t)(rv ? eidx : 0) * FSTRIDE;
    for (int K = tid >> 4; K < Kpad; K += 8) {
      float v = (rv && K < HH) ? (float)rm[K] : 0.f;
      Atile[row * LDSK + K] = (_Float16)v;
    }
  }
  __syncthreads();

  const int wave = tid >> 5;
  const int lane = tid & 31;
  const int col  = (wave << 4) + (lane & 15);
  const int arow = lane & 15;
  const int hi   = lane >> 4;

  v8f acc = {};
  for (int k0 = 0; k0 < Kpad; k0 += 32) {
    v16h a = load_a_frag(&Atile[arow * LDSK + k0 + hi * 8]);
    v16h b = Wfrag[((size_t)(k0 >> 5) * 4 + wave) * 32 + lane];
    acc = __builtin_amdgcn_wmma_f32_16x16x32_f16(false, a, false, b,
                                                 (short)0, acc, false, false);
  }

  float bv = eob[col];
  const int mbase = hi << 3;
#pragma unroll
  for (int r = 0; r < 8; ++r) {
    int eidx = e0 + mbase + r;
    if (eidx < nE) {
      int g = batch[src[eidx]];
      atomicAdd(&sums[(size_t)g * 64 + col], acc[r] + bv);
    }
  }
}

// ---------------------------------------------------------------------------
__global__ void count_kernel(const int* __restrict__ src,
                             const int* __restrict__ batch,
                             float* __restrict__ cnt, int nE) {
  int t = blockIdx.x * blockDim.x + threadIdx.x;
  if (t < nE) atomicAdd(&cnt[batch[src[t]]], 1.0f);
}

__global__ void readout1(const float* __restrict__ sums, const float* __restrict__ cnt,
                         const float* __restrict__ w, const float* __restrict__ b,
                         float* __restrict__ t) {
  int idx = blockIdx.x * blockDim.x + threadIdx.x;
  if (idx >= GG * 64) return;
  int g = idx >> 6, j = idx & 63;
  float c = cnt[g]; c = c > 1.f ? c : 1.f;
  float acc = b[j];
  for (int k = 0; k < 64; ++k) acc += (sums[(g << 6) + k] / c) * w[(k << 6) + j];
  t[idx] = relu_f(acc);
}

__global__ void readout2(const float* __restrict__ t, const float* __restrict__ w,
                         const float* __restrict__ b, float* __restrict__ out) {
  int idx = blockIdx.x * blockDim.x + threadIdx.x;
  if (idx >= GG * 64) return;
  int g = idx >> 6, j = idx & 63;
  float acc = b[j];
  for (int k = 0; k < 64; ++k) acc += t[(g << 6) + k] * w[(k << 6) + j];
  out[idx] = acc;
}

// ---------------------------------------------------------------------------
extern "C" void kernel_launch(void* const* d_in, const int* in_sizes, int n_in,
                              void* d_out, int out_size, void* d_ws, size_t ws_size,
                              hipStream_t stream) {
  (void)in_sizes; (void)n_in; (void)out_size; (void)ws_size;

  const float* x         = (const float*)d_in[0];
  const float* edge_attr = (const float*)d_in[1];
  const int*   eindex    = (const int*)d_in[2];
  const int*   src       = eindex;
  const int*   dst       = eindex + EE;
  const int*   batch     = (const int*)d_in[3];
  const float* enc_xw = (const float*)d_in[4];  const float* enc_xb = (const float*)d_in[5];
  const float* enc_ew = (const float*)d_in[6];  const float* enc_eb = (const float*)d_in[7];
  const float* ew0 = (const float*)d_in[8];   const float* eb0 = (const float*)d_in[9];
  const float* nw0 = (const float*)d_in[10];  const float* nb0 = (const float*)d_in[11];
  const float* ew1 = (const float*)d_in[12];  const float* eb1 = (const float*)d_in[13];
  const float* nw1 = (const float*)d_in[14];  const float* nb1 = (const float*)d_in[15];
  const float* ew2 = (const float*)d_in[16];  const float* eb2 = (const float*)d_in[17];
  const float* nw2 = (const float*)d_in[18];  const float* nb2 = (const float*)d_in[19];
  const float* eow = (const float*)d_in[20];  const float* eob = (const float*)d_in[21];
  const float* r1w = (const float*)d_in[22];  const float* r1b = (const float*)d_in[23];
  const float* r2w = (const float*)d_in[24];  const float* r2b = (const float*)d_in[25];
  float* out = (float*)d_out;

  // ---- workspace carve (256B aligned) ----
  size_t off = 0;
  char* base = (char*)d_ws;
  auto take = [&](size_t bytes) -> char* {
    char* p = base + off;
    off = (off + bytes + 255) & ~(size_t)255;
    return p;
  };
  _Float16* hA  = (_Float16*)take((size_t)NN * FSTRIDE * 2);
  _Float16* hB  = (_Float16*)take((size_t)NN * FSTRIDE * 2);
  float*    agg = (float*)   take((size_t)NN * FSTRIDE * 4);
  _Float16* eA  = (_Float16*)take((size_t)EE * FSTRIDE * 2);
  _Float16* eB  = (_Float16*)take((size_t)EE * FSTRIDE * 2);
  _Float16* wfe0 = (_Float16*)take((size_t)64  * 96 * 2);   // Kpad=64,  NT=6, segW=16
  _Float16* wfn0 = (_Float16*)take((size_t)96  * 96 * 2);   // Kpad=96,  NT=6
  _Float16* wfe1 = (_Float16*)take((size_t)256 * 96 * 2);   // Kpad=256, NT=6, segW=80
  _Float16* wfn1 = (_Float16*)take((size_t)160 * 96 * 2);   // Kpad=160, NT=6
  _Float16* wfe2 = (_Float16*)take((size_t)256 * 96 * 2);
  _Float16* wfn2 = (_Float16*)take((size_t)160 * 96 * 2);
  _Float16* wfeo = (_Float16*)take((size_t)96  * 64 * 2);   // Kpad=96,  NT=4
  float* sums = (float*)take((size_t)GG * 64 * 4);
  float* cnt  = (float*)take((size_t)GG * 4);
  float* tbuf = (float*)take((size_t)GG * 64 * 4);

  const int eGrid64 = (EE + 63) / 64;
  const int eGrid16 = (EE + 15) / 16;
  const int nGrid   = (NN + 15) / 16;

  // ---- encoders ----
  encode_kernel<<<(NN * 16 + 255) / 256, 256, 0, stream>>>(x, enc_xw, enc_xb, hA, NN);
  encode_kernel<<<(EE * 16 + 255) / 256, 256, 0, stream>>>(edge_attr, enc_ew, enc_eb, eA, EE);

  // ---- weight repack into WMMA B-fragment layout ----
  auto conv = [&](const float* W, _Float16* Wf, int hseg, int Nc, int Kpad, int NT,
                  int segW) {
    int total = (Kpad >> 5) * NT * 512;
    conv_wfrag<<<(total + 255) / 256, 256, 0, stream>>>(W, Wf, hseg, Nc, Kpad, NT, segW);
  };
  conv(ew0, wfe0, 16, HH,  64, 6, 16);   // edge L0: 3 segments of 16
  conv(nw0, wfn0, 92, HH,  96, 6, 96);   // node L0: flat K (Kin=92)
  conv(ew1, wfe1, 76, HH, 256, 6, 80);   // edge L1: 3 segments of 80 (76 valid)
  conv(nw1, wfn1, 152, HH, 160, 6, 160); // node L1: flat K (Kin=152)
  conv(ew2, wfe2, 76, HH, 256, 6, 80);
  conv(nw2, wfn2, 152, HH, 160, 6, 160);
  conv(eow, wfeo, 76, 64,  96, 4, 96);   // edge-out: flat K (Kin=76)

  // ---- layer 0 (feature width 16) ----
  hipMemsetAsync(agg, 0, (size_t)NN * FSTRIDE * 4, stream);
  edge_msg_wmma_async<64, 16><<<eGrid64, 192, 0, stream>>>(
      hA, eA, src, dst, (const v16h*)wfe0, eb0, eB, agg, EE);
  node_upd_wmma<<<nGrid, 192, 0, stream>>>(hA, agg, (const v16h*)wfn0, nb0, hB,
                                           NN, 16, 96);
  // ---- layer 1 (width 76) ----
  hipMemsetAsync(agg, 0, (size_t)NN * FSTRIDE * 4, stream);
  edge_msg_wmma_async<256, 80><<<eGrid64, 192, 0, stream>>>(
      hB, eB, src, dst, (const v16h*)wfe1, eb1, eA, agg, EE);
  node_upd_wmma<<<nGrid, 192, 0, stream>>>(hB, agg, (const v16h*)wfn1, nb1, hA,
                                           NN, HH, 160);
  // ---- layer 2 (final h unused downstream: use_nodes=False) ----
  hipMemsetAsync(agg, 0, (size_t)NN * FSTRIDE * 4, stream);
  edge_msg_wmma_async<256, 80><<<eGrid64, 192, 0, stream>>>(
      hA, eA, src, dst, (const v16h*)wfe2, eb2, eB, agg, EE);

  // ---- edge output projection + pooled sums ----
  hipMemsetAsync(sums, 0, (size_t)GG * 64 * 4, stream);
  hipMemsetAsync(cnt,  0, (size_t)GG * 4, stream);
  count_kernel<<<(EE + 255) / 256, 256, 0, stream>>>(src, batch, cnt, EE);
  edge_out_pool_wmma<<<eGrid16, 128, 0, stream>>>(eB, (const v16h*)wfeo, eob,
                                                  src, batch, sums, EE);

  // ---- readout MLP ----
  readout1<<<(GG * 64 + 255) / 256, 256, 0, stream>>>(sums, cnt, r1w, r1b, tbuf);
  readout2<<<(GG * 64 + 255) / 256, 256, 0, stream>>>(tbuf, r2w, r2b, out);
}
